// NeighborhoodCrossAttention_85976655332042
// MI455X (gfx1250) — matrix operand order repaired and checked by reading, not compiled, not run on hardware
//
#include <hip/hip_runtime.h>
#include <hip/hip_bf16.h>

typedef __attribute__((ext_vector_type(16))) _Float16 v16h;
typedef __attribute__((ext_vector_type(8)))  _Float16 h8;
typedef __attribute__((ext_vector_type(8)))  float    v8f;

union AFrag { v16h v; h8 h[2]; _Float16 e[16]; };

#define WMMA_F16(a, b, c) \
  __builtin_amdgcn_wmma_f32_16x16x32_f16(false, (a), false, (b), (short)0, (c), false, false)

// ---------------------------------------------------------------------------
// GEMM:  O[M,N] = A[M,K] * W[N,K]^T + bias[N],  K == 256.
// Block: 256 threads = 8 waves; block tile 64(M) x 64(N).
// K is staged in two 128-wide chunks through LDS (f16, padded rows to kill
// bank conflicts) -> only 4 barriers per block. Each wave owns a 16(M)x32(N)
// strip: 2 accumulators, 16 chained WMMAs, ds_load_b128 fragment reads.
// LDS row stride 136 halves (272B): consecutive fragment lanes land 4 banks
// apart -> 16 lanes x b128 cover all 64 banks conflict-free.
// ---------------------------------------------------------------------------
template<bool A_HALF, bool OUT_HALF>
__global__ __launch_bounds__(256)
void gemm_bias_wmma(const void* __restrict__ Ap, const float* __restrict__ Wp,
                    const float* __restrict__ bias, void* __restrict__ Op,
                    int M, int N)
{
  constexpr int KD  = 256;   // full K
  constexpr int KC  = 128;   // staged chunk
  constexpr int LDP = 136;   // padded LDS row stride (halves)
  __shared__ __align__(16) _Float16 sA[64 * LDP];
  __shared__ __align__(16) _Float16 sB[64 * LDP];

  const int t    = threadIdx.x;
  const int lane = t & 31;
  const int wv   = t >> 5;
  const int mw   = wv & 3;   // 0..3 -> 16-row strip
  const int nw   = wv >> 2;  // 0..1 -> 32-col strip
  const int m0   = blockIdx.x * 64;
  const int n0   = blockIdx.y * 64;

  v8f acc0 = {}, acc1 = {};

  for (int ks = 0; ks < KD; ks += KC) {
    // ---- stage A (64 x 128) and B (64 x 128) chunks into LDS as f16 ----
    // thread t, iter it: row = it*16 + t/16, col = (t%16)*8  (8 elems/thread/iter)
#pragma unroll
    for (int it = 0; it < 4; ++it) {
      const int row = it * 16 + (t >> 4);
      const int col = (t & 15) << 3;
      // A chunk
      if constexpr (A_HALF) {
        const _Float16* a = (const _Float16*)Ap + (size_t)(m0 + row) * KD + ks + col;
        *(h8*)&sA[row * LDP + col] = *(const h8*)a;
      } else {
        const float* a = (const float*)Ap + (size_t)(m0 + row) * KD + ks + col;
        float4 f0 = *(const float4*)a;
        float4 f1 = *(const float4*)(a + 4);
        h8 hv;
        hv[0] = (_Float16)f0.x; hv[1] = (_Float16)f0.y;
        hv[2] = (_Float16)f0.z; hv[3] = (_Float16)f0.w;
        hv[4] = (_Float16)f1.x; hv[5] = (_Float16)f1.y;
        hv[6] = (_Float16)f1.z; hv[7] = (_Float16)f1.w;
        *(h8*)&sA[row * LDP + col] = hv;
      }
      // B chunk (weights always f32, row-major [N, K])
      {
        const float* wr = Wp + (size_t)(n0 + row) * KD + ks + col;
        float4 f0 = *(const float4*)wr;
        float4 f1 = *(const float4*)(wr + 4);
        h8 hv;
        hv[0] = (_Float16)f0.x; hv[1] = (_Float16)f0.y;
        hv[2] = (_Float16)f0.z; hv[3] = (_Float16)f0.w;
        hv[4] = (_Float16)f1.x; hv[5] = (_Float16)f1.y;
        hv[6] = (_Float16)f1.z; hv[7] = (_Float16)f1.w;
        *(h8*)&sB[row * LDP + col] = hv;
      }
    }
    __syncthreads();

    // ---- 4 K-steps of 32, 2 WMMAs each ----
    const int arow = mw * 16 + (lane & 15);
    const int kb   = (lane < 16) ? 0 : 8;   // A-frag K base
    const int kb2  = (lane < 16) ? 0 : 8;   // B-frag K base
    const int bc0  = nw * 32 + (lane & 15);
#pragma unroll
    for (int kc = 0; kc < KC; kc += 32) {
      AFrag af, bf0, bf1;
      af.h[0]  = *(const h8*)&sA[arow * LDP + kc + kb];
      af.h[1]  = *(const h8*)&sA[arow * LDP + kc + kb + 16];
      bf0.h[0] = *(const h8*)&sB[bc0 * LDP + kc + kb2];
      bf0.h[1] = *(const h8*)&sB[bc0 * LDP + kc + kb2 + 16];
      bf1.h[0] = *(const h8*)&sB[(bc0 + 16) * LDP + kc + kb2];
      bf1.h[1] = *(const h8*)&sB[(bc0 + 16) * LDP + kc + kb2 + 16];
      acc0 = WMMA_F16(af.v, bf0.v, acc0);
      acc1 = WMMA_F16(af.v, bf1.v, acc1);
    }
    __syncthreads();
  }

  // ---- epilogue: C layout -> lane n = lane&15, VGPR r -> m = r + (lane<16?0:8)
  const int n   = n0 + nw * 32 + (lane & 15);
  const float bv0 = bias[n];
  const float bv1 = bias[n + 16];
#pragma unroll
  for (int r = 0; r < 8; ++r) {
    const int m = m0 + mw * 16 + r + ((lane < 16) ? 0 : 8);
    const float v0 = acc0[r] + bv0;
    const float v1 = acc1[r] + bv1;
    if constexpr (OUT_HALF) {
      ((_Float16*)Op)[(size_t)m * N + n]      = (_Float16)v0;
      ((_Float16*)Op)[(size_t)m * N + n + 16] = (_Float16)v1;
    } else {
      ((float*)Op)[(size_t)m * N + n]      = v0;
      ((float*)Op)[(size_t)m * N + n + 16] = v1;
    }
  }
}

// ---------------------------------------------------------------------------
// Neighborhood attention. One wave = (b, head, row i, 16-query j-strip).
// All 16 queries share the same 7 key-rows; their clamped column windows fit
// in a 22-column strip -> 154 keys, padded to 160 (10 B-tiles of 16).
// QK^T: 10 WMMAs. Bias+mask+softmax in VALU. P -> LDS -> A-frags, V staged
// transposed in LDS. PV: 10 WMMAs (hd=32 = two N groups).
// ---------------------------------------------------------------------------
__global__ __launch_bounds__(128)
void nca_attention(const float* __restrict__ q_extra, const _Float16* __restrict__ kv,
                   const float* __restrict__ rpb, _Float16* __restrict__ aout)
{
  constexpr float SCALE = 0.17677669529663687f;  // 32^-0.5
  __shared__ __align__(16) _Float16 sP[4][16 * 160];   // softmaxed probs per wave
  __shared__ __align__(16) _Float16 sVt[4][32 * 160];  // V transposed (hd-major) per wave

  const int lane = threadIdx.x & 31;
  const int w    = threadIdx.x >> 5;
  const int u    = blockIdx.x * 4 + w;   // 0 .. 3583
  const int j0tab[4] = {0, 16, 32, 40};
  const int j0 = j0tab[u & 3];
  const int i  = (u >> 2) % 56;
  const int h  = (u / 224) % 8;
  const int b  = u / 1792;

  const int rstart  = min(max(i - 3, 0), 49);   // shared row-window start
  const int c0      = min(max(j0 - 3, 0), 34);  // 22-column key strip start
  const int posBase = b * 3136;

  // ---- Q fragment: A layout, scaled, f32 -> f16 ----
  AFrag qa;
  {
    const int mq = lane & 15;
    const int jq = j0 + mq;
    const float* qp = q_extra + (size_t)(posBase + i * 56 + jq) * 256 + h * 32;
    const int kb = (lane < 16) ? 0 : 8;
    float4 A0 = *(const float4*)(qp + kb);
    float4 A1 = *(const float4*)(qp + kb + 4);
    float4 A2 = *(const float4*)(qp + kb + 16);
    float4 A3 = *(const float4*)(qp + kb + 20);
    float tmp[16] = {A0.x, A0.y, A0.z, A0.w, A1.x, A1.y, A1.z, A1.w,
                     A2.x, A2.y, A2.z, A2.w, A3.x, A3.y, A3.z, A3.w};
#pragma unroll
    for (int q = 0; q < 16; ++q) qa.e[q] = (_Float16)(tmp[q] * SCALE);
  }

  // ---- scores: 10 WMMAs, K loaded straight from global in B layout ----
  v8f sc[10];
#pragma unroll
  for (int tb = 0; tb < 10; ++tb) {
    int tk = tb * 16 + (lane & 15);
    if (tk > 153) tk = 153;                     // clamp pad keys (masked later)
    const int kr = tk / 22, kcc = tk - kr * 22;
    const int pos = posBase + (rstart + kr) * 56 + (c0 + kcc);
    const _Float16* kp = kv + (size_t)pos * 512 + h * 32 + ((lane < 16) ? 0 : 8);
    AFrag bf;
    bf.h[0] = *(const h8*)kp;
    bf.h[1] = *(const h8*)(kp + 16);
    v8f z = {};
    sc[tb] = WMMA_F16(qa.v, bf.v, z);
  }

  // ---- relative position bias + window mask ----
  const int pbi = 3 + max(3 - i, 0) + min(52 - i, 0);
#pragma unroll
  for (int tb = 0; tb < 10; ++tb) {
    const int tkr = tb * 16 + (lane & 15);
    const bool pad = tkr > 153;
    const int tc  = pad ? 153 : tkr;
    const int kr  = tc / 22;
    const int cabs = c0 + (tc - kr * 22);
#pragma unroll
    for (int r = 0; r < 8; ++r) {
      const int m  = r + ((lane < 16) ? 0 : 8);
      const int j  = j0 + m;
      const int cs = min(max(j - 3, 0), 49);
      const bool in = !pad && (cabs >= cs) && (cabs <= cs + 6);
      if (in) {
        const int pbj = 3 + max(3 - j, 0) + min(52 - j, 0);
        sc[tb][r] += rpb[(h * 13 + (pbi + kr)) * 13 + (pbj + (cabs - cs))];
      } else {
        sc[tb][r] = -1e30f;
      }
    }
  }

  // ---- softmax per query row (16-lane half-wave + register reduction) ----
#pragma unroll
  for (int r = 0; r < 8; ++r) {
    float mx = -1e30f;
#pragma unroll
    for (int tb = 0; tb < 10; ++tb) mx = fmaxf(mx, sc[tb][r]);
#pragma unroll
    for (int x = 8; x >= 1; x >>= 1) mx = fmaxf(mx, __shfl_xor(mx, x, 32));
    float sm = 0.f;
#pragma unroll
    for (int tb = 0; tb < 10; ++tb) {
      float e = __expf(sc[tb][r] - mx);
      sc[tb][r] = e;
      sm += e;
    }
#pragma unroll
    for (int x = 8; x >= 1; x >>= 1) sm += __shfl_xor(sm, x, 32);
    const float inv = 1.0f / sm;
    const int m = r + ((lane < 16) ? 0 : 8);
#pragma unroll
    for (int tb = 0; tb < 10; ++tb)
      sP[w][m * 160 + tb * 16 + (lane & 15)] = (_Float16)(sc[tb][r] * inv);
  }

  // ---- stage V transposed into LDS (hd-major rows of length 160) ----
#pragma unroll
  for (int tt = 0; tt < 5; ++tt) {
    const int tk = lane + tt * 32;
    const int tc = min(tk, 153);
    const int kr = tc / 22, kcc = tc - kr * 22;
    const int pos = posBase + (rstart + kr) * 56 + (c0 + kcc);
    const _Float16* vp = kv + (size_t)pos * 512 + 256 + h * 32;
    _Float16 ve[32];
    *(h8*)&ve[0]  = *(const h8*)vp;
    *(h8*)&ve[8]  = *(const h8*)(vp + 8);
    *(h8*)&ve[16] = *(const h8*)(vp + 16);
    *(h8*)&ve[24] = *(const h8*)(vp + 24);
#pragma unroll
    for (int d = 0; d < 32; ++d) sVt[w][d * 160 + tk] = ve[d];
  }

  // wave-private LDS handoff: drain DS ops before cross-lane reads
  asm volatile("s_wait_dscnt 0" ::: "memory");

  // ---- PV: O[16 x 32] = P[16 x 160] @ V[160 x 32] -> 5 K-chunks x 2 N groups ----
  v8f o0 = {}, o1 = {};
#pragma unroll
  for (int ch = 0; ch < 5; ++ch) {
    AFrag pa, vb0, vb1;
    const int pm  = lane & 15;
    const int kb3 = (lane < 16) ? 0 : 8;
    const _Float16* pb = &sP[w][pm * 160 + ch * 32 + kb3];
    pa.h[0] = *(const h8*)pb;
    pa.h[1] = *(const h8*)(pb + 16);
    const int kb4 = (lane < 16) ? 0 : 8;
    const _Float16* vb = &sVt[w][(lane & 15) * 160 + ch * 32 + kb4];
    vb0.h[0] = *(const h8*)vb;
    vb0.h[1] = *(const h8*)(vb + 16);
    const _Float16* vb2 = &sVt[w][((lane & 15) + 16) * 160 + ch * 32 + kb4];
    vb1.h[0] = *(const h8*)vb2;
    vb1.h[1] = *(const h8*)(vb2 + 16);
    o0 = WMMA_F16(pa.v, vb0.v, o0);
    o1 = WMMA_F16(pa.v, vb1.v, o1);
  }

  // ---- write attention output (f16, feeds final projection GEMM) ----
  const int n = lane & 15;
#pragma unroll
  for (int r = 0; r < 8; ++r) {
    const int m = r + ((lane < 16) ? 0 : 8);
    const int pos = posBase + i * 56 + j0 + m;
    _Float16* op = aout + (size_t)pos * 256 + h * 32;
    op[n]      = (_Float16)o0[r];
    op[n + 16] = (_Float16)o1[r];
  }
}

// ---------------------------------------------------------------------------
extern "C" void kernel_launch(void* const* d_in, const int* in_sizes, int n_in,
                              void* d_out, int out_size, void* d_ws, size_t ws_size,
                              hipStream_t stream)
{
  (void)in_sizes; (void)n_in; (void)out_size; (void)ws_size;
  const float* x       = (const float*)d_in[0];
  const float* q_extra = (const float*)d_in[1];
  const float* kv_w    = (const float*)d_in[2];
  const float* kv_b    = (const float*)d_in[3];
  const float* rpb     = (const float*)d_in[4];
  const float* proj_w  = (const float*)d_in[5];
  const float* proj_b  = (const float*)d_in[6];
  float* out = (float*)d_out;

  _Float16* kvbuf   = (_Float16*)d_ws;                 // [6272, 512] f16 (6.4 MB)
  _Float16* attnout = kvbuf + (size_t)6272 * 512;      // [6272, 256] f16 (3.2 MB)

  const int M = 6272;  // B*H*W

  // 1) KV projection: (6272x256) @ (256x512)^T + kv_b -> f16
  gemm_bias_wmma<false, true><<<dim3(98, 8), 256, 0, stream>>>(
      x, kv_w, kv_b, kvbuf, M, 512);

  // 2) neighborhood attention: 3584 wave-units / 4 waves per block
  nca_attention<<<896, 128, 0, stream>>>(q_extra, kvbuf, rpb, attnout);

  // 3) output projection: (6272x256) @ (256x256)^T + proj_b -> f32
  gemm_bias_wmma<true, false><<<dim3(98, 4), 256, 0, stream>>>(
      attnout, proj_w, proj_b, out, M, 256);
}
